// CentralAttention1_2173253452558
// MI455X (gfx1250) — matrix operand order repaired and checked by reading, not compiled, not run on hardware
//
#include <hip/hip_runtime.h>

typedef __attribute__((ext_vector_type(16))) __bf16 v16bf;
typedef __attribute__((ext_vector_type(8)))  float  v8f;

union Frag16 { v16bf v; unsigned u[8]; };

// ---------------------------------------------------------------------------
// Generic WMMA bf16 GEMM:  C(M,N) = act( A(M,K) * B(N,K)^T + bias )
// Tile: 128x128 per block (256 threads = 8 waves), K chunk = 32.
// Double-buffered LDS staging (one barrier per chunk, global latency hidden
// under WMMA work). Wave tiling 2x4: each wave owns 32 rows x 64 cols ->
// 2 A-frags + 4 B-frags (12 ds_load_b128) per 8 WMMAs.
// B is stored row-major [N][K] (i.e. weight rows directly).
// act: 0=none, 1=relu, 2=leaky(0.01)
// ---------------------------------------------------------------------------
__global__ __launch_bounds__(256) void gemm_bf16_kernel(
    const __bf16* __restrict__ A, int lda,
    const __bf16* __restrict__ B,              // [N][K]
    const float*  __restrict__ bias,           // [N] or null
    float*  __restrict__ Cf,                   // f32 out or null
    __bf16* __restrict__ Cb,                   // bf16 out or null
    int ldc, int M, int N, int K, int act)
{
    __shared__ unsigned short As[2][128 * 32];
    __shared__ unsigned short Bs[2][128 * 32];

    const int tid  = threadIdx.x;
    const int wave = tid >> 5;
    const int lane = tid & 31;
    const int r16  = lane & 15;
    const int hi   = lane >> 4;
    const int wr   = wave & 3;        // row group: 32 rows
    const int wc   = wave >> 2;       // col group: 64 cols
    const int tm   = blockIdx.x * 128;
    const int tn   = blockIdx.y * 128;

    v8f acc[2][4];
    #pragma unroll
    for (int m2 = 0; m2 < 2; m2++)
        #pragma unroll
        for (int ns = 0; ns < 4; ns++)
            #pragma unroll
            for (int j = 0; j < 8; j++) acc[m2][ns][j] = 0.f;

    const int lrow = tid >> 1;           // 0..127
    const int lcol = (tid & 1) * 16;     // 0 or 16 (elements)
    const __bf16* ag = A + (size_t)(tm + lrow) * lda + lcol;
    const __bf16* bg = B + (size_t)(tn + lrow) * K   + lcol;

    // prologue: stage chunk 0 into buffer 0
    uint4 ra0 = *(const uint4*)ag;
    uint4 ra1 = *(const uint4*)(ag + 8);
    uint4 rb0 = *(const uint4*)bg;
    uint4 rb1 = *(const uint4*)(bg + 8);
    *(uint4*)&As[0][lrow * 32 + lcol]     = ra0;
    *(uint4*)&As[0][lrow * 32 + lcol + 8] = ra1;
    *(uint4*)&Bs[0][lrow * 32 + lcol]     = rb0;
    *(uint4*)&Bs[0][lrow * 32 + lcol + 8] = rb1;
    __syncthreads();

    const int nchunks = K >> 5;
    for (int ic = 0; ic < nchunks; ic++) {
        const int cur = ic & 1;
        const bool more = (ic + 1) < nchunks;
        if (more) {   // issue next chunk's global loads; they complete under WMMA
            const __bf16* agn = ag + (ic + 1) * 32;
            const __bf16* bgn = bg + (ic + 1) * 32;
            ra0 = *(const uint4*)agn;
            ra1 = *(const uint4*)(agn + 8);
            rb0 = *(const uint4*)bgn;
            rb1 = *(const uint4*)(bgn + 8);
        }

        // A fragments: lane holds row r16; VGPR v: k0 = 16*(v/4)+2*(v%4)+8*hi
        Frag16 af[2];
        #pragma unroll
        for (int m2 = 0; m2 < 2; m2++)
            #pragma unroll
            for (int v = 0; v < 8; v++) {
                int k0 = ((v >> 2) << 4) + ((v & 3) << 1) + (hi << 3);
                af[m2].u[v] = *(const unsigned*)&As[cur][(wr * 32 + m2 * 16 + r16) * 32 + k0];
            }
        // B fragments: lane holds col r16; VGPR v: k0 = 16*(v/4)+2*(v%4)+8*hi
        Frag16 bfr[4];
        #pragma unroll
        for (int ns = 0; ns < 4; ns++)
            #pragma unroll
            for (int v = 0; v < 8; v++) {
                int k0 = ((v >> 2) << 4) + ((v & 3) << 1) + (hi << 3);
                bfr[ns].u[v] = *(const unsigned*)&Bs[cur][(wc * 64 + ns * 16 + r16) * 32 + k0];
            }
        #pragma unroll
        for (int m2 = 0; m2 < 2; m2++)
            #pragma unroll
            for (int ns = 0; ns < 4; ns++)
                acc[m2][ns] = __builtin_amdgcn_wmma_f32_16x16x32_bf16(
                    false, af[m2].v, false, bfr[ns].v, (short)0, acc[m2][ns], false, false);

        if (more) {
            const int nxt = cur ^ 1;
            *(uint4*)&As[nxt][lrow * 32 + lcol]     = ra0;
            *(uint4*)&As[nxt][lrow * 32 + lcol + 8] = ra1;
            *(uint4*)&Bs[nxt][lrow * 32 + lcol]     = rb0;
            *(uint4*)&Bs[nxt][lrow * 32 + lcol + 8] = rb1;
            __syncthreads();
        }
    }

    #pragma unroll
    for (int ns = 0; ns < 4; ns++) {
        int col = tn + wc * 64 + ns * 16 + r16;
        float bval = bias ? bias[col] : 0.f;
        #pragma unroll
        for (int m2 = 0; m2 < 2; m2++)
            #pragma unroll
            for (int v = 0; v < 8; v++) {
                int row = tm + wr * 32 + m2 * 16 + v + (hi << 3);
                float x = acc[m2][ns][v] + bval;
                if (act == 1) x = fmaxf(x, 0.f);
                else if (act == 2) x = (x >= 0.f) ? x : 0.01f * x;
                if (row < M && col < N) {
                    if (Cf) Cf[(size_t)row * ldc + col] = x;
                    if (Cb) Cb[(size_t)row * ldc + col] = (__bf16)x;
                }
            }
    }
}

// ---------------------------------------------------------------------------
// Fused conv1(5) + relu + conv2(3, via WMMA) + relu -> feature row [8000] bf16
// One block per row. conv1 in VALU (5 MACs/out) into LDS bf16 [32][256],
// conv2 as GEMM: M=256 positions (250 valid), N=32 channels, K=96 (ci*3+kw).
// feature index = ch*250 + p  (matches x.reshape(N, 32*250)).
// ---------------------------------------------------------------------------
__global__ __launch_bounds__(256) void conv_feat_kernel(
    const float* __restrict__ scan,            // [rows][256]
    const float* __restrict__ w1,              // [32][5]
    const float* __restrict__ b1,              // [32]
    const __bf16* __restrict__ w2b,            // [32][96] bf16
    const float* __restrict__ b2,              // [32]
    __bf16* __restrict__ feat)                 // [rows][8000]
{
    __shared__ float ss[256];
    __shared__ unsigned short c1[32 * 256];    // conv1 out bf16, [ci][pp]

    const int row = blockIdx.x;
    const int tid = threadIdx.x;
    ss[tid] = scan[(size_t)row * 256 + tid];
    __syncthreads();

    for (int i = tid; i < 32 * 256; i += 256) {
        int ci = i >> 8, pp = i & 255;
        float a = 0.f;
        if (pp < 252) {
            a = b1[ci];
            #pragma unroll
            for (int kw = 0; kw < 5; kw++) a += ss[pp + kw] * w1[ci * 5 + kw];
            a = fmaxf(a, 0.f);
        }
        __bf16 h = (__bf16)a;
        c1[i] = *(unsigned short*)&h;
    }
    __syncthreads();

    const int wave = tid >> 5, lane = tid & 31;
    const int r16 = lane & 15, hi = lane >> 4;

    v8f acc[2][2];
    for (int m2 = 0; m2 < 2; m2++)
        for (int n2 = 0; n2 < 2; n2++)
            #pragma unroll
            for (int j = 0; j < 8; j++) acc[m2][n2][j] = 0.f;

    for (int kc = 0; kc < 96; kc += 32) {
        Frag16 bf_[2];
        #pragma unroll
        for (int n2 = 0; n2 < 2; n2++) {
            int ch = n2 * 16 + r16;
            #pragma unroll
            for (int v = 0; v < 8; v++) {
                int k0 = kc + ((v >> 2) << 4) + ((v & 3) << 1) + (hi << 3);
                bf_[n2].u[v] = *(const unsigned*)&w2b[ch * 96 + k0]; // pairs contiguous
            }
        }
        #pragma unroll
        for (int m2 = 0; m2 < 2; m2++) {
            int p = (wave * 2 + m2) * 16 + r16;
            Frag16 af;
            #pragma unroll
            for (int v = 0; v < 8; v++) {
                int k0 = kc + ((v >> 2) << 4) + ((v & 3) << 1) + (hi << 3);
                int k1 = k0 + 1;
                int ci0 = k0 / 3, kw0 = k0 - ci0 * 3;
                int ci1 = k1 / 3, kw1 = k1 - ci1 * 3;
                int p0 = p + kw0; if (p0 > 255) p0 = 255;
                int p1 = p + kw1; if (p1 > 255) p1 = 255;
                unsigned e0 = c1[ci0 * 256 + p0];
                unsigned e1 = c1[ci1 * 256 + p1];
                af.u[v] = e0 | (e1 << 16);
            }
            #pragma unroll
            for (int n2 = 0; n2 < 2; n2++)
                acc[m2][n2] = __builtin_amdgcn_wmma_f32_16x16x32_bf16(
                    false, af.v, false, bf_[n2].v, (short)0, acc[m2][n2], false, false);
        }
    }

    for (int m2 = 0; m2 < 2; m2++)
        for (int n2 = 0; n2 < 2; n2++) {
            int ch = n2 * 16 + r16;
            float bb = b2[ch];
            #pragma unroll
            for (int v = 0; v < 8; v++) {
                int p = (wave * 2 + m2) * 16 + v + (hi << 3);
                if (p < 250) {
                    float x = fmaxf(acc[m2][n2][v] + bb, 0.f);
                    feat[(size_t)row * 8000 + ch * 250 + p] = (__bf16)x;
                }
            }
        }
}

// --------------------------- small VALU kernels ----------------------------
__global__ void cvt_kernel(const float* __restrict__ src, __bf16* __restrict__ dst, int n) {
    int i = blockIdx.x * 256 + threadIdx.x;
    if (i < n) dst[i] = (__bf16)src[i];
}

// fc2 (256->10) + concat [obs(48), acts(2), feats(10)] -> inps[row][60]
__global__ void fc2_concat_kernel(const float* __restrict__ obs, const float* __restrict__ acts,
                                  const float* __restrict__ hidden,      // [rows][256]
                                  const float* __restrict__ fc_w2, const float* __restrict__ fc_b2,
                                  float* __restrict__ inps)
{
    int idx = blockIdx.x * 256 + threadIdx.x;
    int row = idx >> 6, slot = idx & 63;
    if (row >= 12288 || slot >= 60) return;
    float v;
    if (slot < 48) v = obs[(size_t)row * 48 + slot];
    else if (slot < 50) v = acts[(size_t)row * 2 + (slot - 48)];
    else {
        int j = slot - 50;
        float a = fc_b2[j];
        const float* h = hidden + (size_t)row * 256;
        const float* w = fc_w2 + j * 256;
        for (int i = 0; i < 256; i++) a += h[i] * w[i];
        v = a;
    }
    inps[(size_t)row * 60 + slot] = v;
}

// per-feature mean / rsqrt(var+eps) over 12288 rows; one block per feature
__global__ __launch_bounds__(256) void stats_kernel(const float* __restrict__ inps,
                                                    float* __restrict__ mu, float* __restrict__ rstd)
{
    __shared__ float s1[256], s2[256];
    int f = blockIdx.x, tid = threadIdx.x;
    float a = 0.f, b = 0.f;
    for (int r = tid; r < 12288; r += 256) {
        float x = inps[(size_t)r * 60 + f];
        a += x; b += x * x;
    }
    s1[tid] = a; s2[tid] = b; __syncthreads();
    for (int s = 128; s > 0; s >>= 1) {
        if (tid < s) { s1[tid] += s1[tid + s]; s2[tid] += s2[tid + s]; }
        __syncthreads();
    }
    if (tid == 0) {
        float m = s1[0] * (1.f / 12288.f);
        float var = s2[0] * (1.f / 12288.f) - m * m;
        mu[f] = m; rstd[f] = rsqrtf(var + 1e-5f);
    }
}

// normalize + encoder(60->128) + leaky -> emb[row][0:128] (bf16, lda 256)
__global__ __launch_bounds__(128) void enc_kernel(const float* __restrict__ inps,
                                                  const float* __restrict__ mu, const float* __restrict__ rstd,
                                                  const float* __restrict__ enc_w, const float* __restrict__ enc_b,
                                                  __bf16* __restrict__ emb)
{
    __shared__ float ns[60];
    int row = blockIdx.x, tid = threadIdx.x;
    if (tid < 60) ns[tid] = (inps[(size_t)row * 60 + tid] - mu[tid]) * rstd[tid];
    __syncthreads();
    float a = enc_b[tid];
    const float* w = enc_w + tid * 60;
    for (int i = 0; i < 60; i++) a += ns[i] * w[i];
    a = (a >= 0.f) ? a : 0.01f * a;
    emb[(size_t)row * 256 + tid] = (__bf16)a;
}

// cross-agent attention: softmax over the 2 non-self agents, per (head,b).
// writes emb[row][128:256]
__global__ __launch_bounds__(256) void attn_kernel(const float* __restrict__ qb,
                                                   const float* __restrict__ kb,
                                                   const float* __restrict__ vb,
                                                   __bf16* __restrict__ emb)
{
    int idx = blockIdx.x * 256 + threadIdx.x;      // 8 heads * 4096
    int n = idx >> 12, b = idx & 4095;
    float qv[3][16], kv[3][16], vv[3][16];
    for (int g = 0; g < 3; g++) {
        size_t base = ((size_t)(g * 4096 + b)) * 128 + n * 16;
        #pragma unroll
        for (int d = 0; d < 16; d++) {
            qv[g][d] = qb[base + d]; kv[g][d] = kb[base + d]; vv[g][d] = vb[base + d];
        }
    }
    for (int g = 0; g < 3; g++) {
        float lg[3];
        #pragma unroll
        for (int o = 0; o < 3; o++) {
            float s = 0.f;
            #pragma unroll
            for (int d = 0; d < 16; d++) s += qv[g][d] * kv[o][d];
            lg[o] = s * 0.25f;                      // 1/sqrt(16)
        }
        int o1 = (g == 0) ? 1 : 0;
        int o2 = (g == 2) ? 1 : 2;
        float m = fmaxf(lg[o1], lg[o2]);
        float e1 = __expf(lg[o1] - m), e2 = __expf(lg[o2] - m);
        float inv = 1.f / (e1 + e2);
        float w1 = e1 * inv, w2 = e2 * inv;
        size_t ebase = ((size_t)(g * 4096 + b)) * 256 + 128 + n * 16;
        #pragma unroll
        for (int d = 0; d < 16; d++)
            emb[ebase + d] = (__bf16)(w1 * vv[o1][d] + w2 * vv[o2][d]);
    }
}

// final head: 256 -> 1
__global__ void head_kernel(const __bf16* __restrict__ h2, const float* __restrict__ w3,
                            const float* __restrict__ b3, float* __restrict__ out)
{
    int row = blockIdx.x * 256 + threadIdx.x;
    if (row >= 12288) return;
    float a = b3[0];
    const __bf16* h = h2 + (size_t)row * 256;
    for (int i = 0; i < 256; i++) a += (float)h[i] * w3[i];
    out[row] = a;
}

// ---------------------------------------------------------------------------
extern "C" void kernel_launch(void* const* d_in, const int* in_sizes, int n_in,
                              void* d_out, int out_size, void* d_ws, size_t ws_size,
                              hipStream_t stream)
{
    const float* obs   = (const float*)d_in[0];
    const float* acts  = (const float*)d_in[1];
    const float* scan  = (const float*)d_in[2];
    const float* cw1   = (const float*)d_in[3];
    const float* cb1   = (const float*)d_in[4];
    const float* cw2   = (const float*)d_in[5];
    const float* cb2   = (const float*)d_in[6];
    const float* fw1   = (const float*)d_in[7];
    const float* fb1   = (const float*)d_in[8];
    const float* fw2   = (const float*)d_in[9];
    const float* fb2   = (const float*)d_in[10];
    const float* encw  = (const float*)d_in[11];
    const float* encb  = (const float*)d_in[12];
    const float* Wk    = (const float*)d_in[13];
    const float* Wv    = (const float*)d_in[14];
    const float* bv    = (const float*)d_in[15];
    const float* Wq    = (const float*)d_in[16];
    const float* q1w1  = (const float*)d_in[17];
    const float* q1b1  = (const float*)d_in[18];
    const float* q1w2  = (const float*)d_in[19];
    const float* q1b2  = (const float*)d_in[20];
    const float* q1w3  = (const float*)d_in[21];
    const float* q1b3  = (const float*)d_in[22];
    const float* q2w1  = (const float*)d_in[23];
    const float* q2b1  = (const float*)d_in[24];
    const float* q2w2  = (const float*)d_in[25];
    const float* q2b2  = (const float*)d_in[26];
    const float* q2w3  = (const float*)d_in[27];
    const float* q2b3  = (const float*)d_in[28];

    const int ROWS = 12288;
    char* wsb = (char*)d_ws;
    size_t off = 0;
    auto alloc = [&](size_t bytes) -> void* {
        void* p = wsb + off;
        off = (off + bytes + 255) & ~(size_t)255;
        return p;
    };

    __bf16* feat   = (__bf16*)alloc((size_t)ROWS * 8000 * 2);   // conv features
    __bf16* fw1b   = (__bf16*)alloc((size_t)8000 * 256 * 2);
    __bf16* cw2b   = (__bf16*)alloc(3072 * 2);
    __bf16* wkb    = (__bf16*)alloc(16384 * 2);
    __bf16* wvb    = (__bf16*)alloc(16384 * 2);
    __bf16* wqb    = (__bf16*)alloc(49152 * 2);
    __bf16* m11    = (__bf16*)alloc(65536 * 2);
    __bf16* m12    = (__bf16*)alloc(65536 * 2);
    __bf16* m21    = (__bf16*)alloc(65536 * 2);
    __bf16* m22    = (__bf16*)alloc(65536 * 2);
    float*  hidden = (float*) alloc((size_t)ROWS * 256 * 4);
    float*  inps   = (float*) alloc((size_t)ROWS * 60 * 4);
    float*  mu     = (float*) alloc(64 * 4);
    float*  rstd   = (float*) alloc(64 * 4);
    __bf16* emb    = (__bf16*)alloc((size_t)ROWS * 256 * 2);
    float*  keys   = (float*) alloc((size_t)ROWS * 128 * 4);
    float*  vals   = (float*) alloc((size_t)ROWS * 128 * 4);
    float*  qbuf   = (float*) alloc((size_t)ROWS * 128 * 4);
    __bf16* h1     = (__bf16*)alloc((size_t)ROWS * 256 * 2);
    __bf16* h2     = (__bf16*)alloc((size_t)ROWS * 256 * 2);

    float* out = (float*)d_out;

    auto cvt = [&](const float* s, __bf16* d, int n) {
        cvt_kernel<<<(n + 255) / 256, 256, 0, stream>>>(s, d, n);
    };
    cvt(fw1, fw1b, 8000 * 256);
    cvt(cw2, cw2b, 3072);
    cvt(Wk, wkb, 16384);
    cvt(Wv, wvb, 16384);
    cvt(Wq, wqb, 49152);
    cvt(q1w1, m11, 65536);
    cvt(q1w2, m12, 65536);
    cvt(q2w1, m21, 65536);
    cvt(q2w2, m22, 65536);

    // conv1 + conv2 (WMMA) -> features
    conv_feat_kernel<<<ROWS, 256, 0, stream>>>(scan, cw1, cb1, cw2b, cb2, feat);

    // fc1: 12288x8000 @ (256x8000)^T, relu -> hidden f32
    gemm_bf16_kernel<<<dim3(ROWS / 128, 2), 256, 0, stream>>>(
        feat, 8000, fw1b, fb1, hidden, (__bf16*)nullptr, 256, ROWS, 256, 8000, 1);

    // fc2 + concat -> inps[12288][60]
    fc2_concat_kernel<<<(ROWS * 64) / 256, 256, 0, stream>>>(obs, acts, hidden, fw2, fb2, inps);

    // batch stats
    stats_kernel<<<60, 256, 0, stream>>>(inps, mu, rstd);

    // normalize + encoder -> emb[:, :128]
    enc_kernel<<<ROWS, 128, 0, stream>>>(inps, mu, rstd, encw, encb, emb);

    // keys = sa @ Wk^T   (Wk flattened is [n*16+d][128] = [N][K])
    gemm_bf16_kernel<<<dim3(ROWS / 128, 1), 256, 0, stream>>>(
        emb, 256, wkb, (const float*)nullptr, keys, (__bf16*)nullptr, 128, ROWS, 128, 128, 0);
    // vals = leaky(sa @ Wv^T + bv)
    gemm_bf16_kernel<<<dim3(ROWS / 128, 1), 256, 0, stream>>>(
        emb, 256, wvb, bv, vals, (__bf16*)nullptr, 128, ROWS, 128, 128, 2);
    // q: per-group weight
    for (int g = 0; g < 3; g++) {
        gemm_bf16_kernel<<<dim3(4096 / 128, 1), 256, 0, stream>>>(
            emb + (size_t)g * 4096 * 256, 256, wqb + (size_t)g * 16384,
            (const float*)nullptr, qbuf + (size_t)g * 4096 * 128, (__bf16*)nullptr,
            128, 4096, 128, 128, 0);
    }

    // attention combine -> emb[:, 128:256]
    attn_kernel<<<(8 * 4096) / 256, 256, 0, stream>>>(qbuf, keys, vals, emb);

    // MLP heads
    gemm_bf16_kernel<<<dim3(ROWS / 128, 2), 256, 0, stream>>>(
        emb, 256, m11, q1b1, (float*)nullptr, h1, 256, ROWS, 256, 256, 1);
    gemm_bf16_kernel<<<dim3(ROWS / 128, 2), 256, 0, stream>>>(
        h1, 256, m12, q1b2, (float*)nullptr, h2, 256, ROWS, 256, 256, 1);
    head_kernel<<<ROWS / 256, 256, 0, stream>>>(h2, q1w3, q1b3, out);

    gemm_bf16_kernel<<<dim3(ROWS / 128, 2), 256, 0, stream>>>(
        emb, 256, m21, q2b1, (float*)nullptr, h1, 256, ROWS, 256, 256, 1);
    gemm_bf16_kernel<<<dim3(ROWS / 128, 2), 256, 0, stream>>>(
        h1, 256, m22, q2b2, (float*)nullptr, h2, 256, ROWS, 256, 256, 1);
    head_kernel<<<ROWS / 256, 256, 0, stream>>>(h2, q2w3, q2b3, out + ROWS);

    (void)in_sizes; (void)n_in; (void)out_size; (void)ws_size;
}
